// MLA_64132451664699
// MI455X (gfx1250) — matrix operand order repaired and checked by reading, not compiled, not run on hardware
//
#include <hip/hip_runtime.h>
#include <hip/hip_bf16.h>
#include <cstdint>

// ---------------- problem constants (match reference) ----------------
#define B_    2
#define S_    2048
#define D_    2048
#define H_    16
#define DH_   384
#define RD_   192
#define DCQ_  4608
#define DCKV_ 1536
#define HD_   (H_*DH_)    // 6144
#define HR_   (H_*RD_)    // 3072
#define QKD_  (DH_+RD_)   // 576
#define MTOK  (B_*S_)     // 4096 token rows

// ---------------- types ----------------
typedef __bf16 bf16;
typedef __attribute__((ext_vector_type(16))) __bf16 bf16x16;
typedef __attribute__((ext_vector_type(8)))  __bf16 bf16x8;
typedef __attribute__((ext_vector_type(8)))  float  floatx8;

__device__ __forceinline__ bf16 f2bf(float f) {
  unsigned u = __builtin_bit_cast(unsigned, f);
  unsigned r = u + 0x7FFFu + ((u >> 16) & 1u);          // round-to-nearest-even
  unsigned short h = (unsigned short)(r >> 16);
  return __builtin_bit_cast(bf16, h);
}
__device__ __forceinline__ float bf2f(bf16 b) {
  unsigned short h = __builtin_bit_cast(unsigned short, b);
  unsigned u = ((unsigned)h) << 16;
  return __builtin_bit_cast(float, u);
}
__device__ __forceinline__ floatx8 zero8() {
  floatx8 z = {0.f,0.f,0.f,0.f,0.f,0.f,0.f,0.f};
  return z;
}
// Build a 16xbf16 fragment (8 VGPRs) from two 16B chunks.
__device__ __forceinline__ bf16x16 ldfrag(const bf16* p0, const bf16* p1) {
  bf16x8 a = *(const bf16x8*)p0;
  bf16x8 b = *(const bf16x8*)p1;
  bf16x16 r;
#pragma unroll
  for (int i = 0; i < 8; ++i) { r[i] = a[i]; r[i+8] = b[i]; }
  return r;
}
__device__ __forceinline__ floatx8 wmma_bf(bf16x16 a, bf16x16 b, floatx8 c) {
  // D = A(16x32) * B(32x16) + C, f32 accumulate
  return __builtin_amdgcn_wmma_f32_16x16x32_bf16(false, a, false, b, (short)0, c, false, false);
}

// ---- gfx1250 async global->LDS copy (ASYNCcnt-tracked DMA into LDS) ----
// Low 32 bits of a generic pointer into __shared__ are the workgroup-relative
// LDS byte offset (flat LDS aperture truncates to addr[31:0]).
__device__ __forceinline__ void async_b128(void* lds, const void* gaddr) {
  unsigned loff = (unsigned)(uintptr_t)lds;
  asm volatile("global_load_async_to_lds_b128 %0, %1, off"
               :: "v"(loff), "v"(gaddr) : "memory");
}
template <int NCNT>
__device__ __forceinline__ void wait_async() {
  asm volatile("s_wait_asynccnt %0" :: "i"(NCNT) : "memory");
}

// ---------------- f32 -> bf16 conversion ----------------
__global__ __launch_bounds__(256) void cvt_f32_bf16(const float* __restrict__ in,
                                                    bf16* __restrict__ out, long n) {
  long i = (long)blockIdx.x * 256 + threadIdx.x;
  if (i < n) out[i] = f2bf(in[i]);
}

// ---------------- generic GEMM: C[M,N] = A[M,K](bf16) * W[N,K]^T(bf16) ----------------
// block tile 128x128, 8 waves, wave tile 32x64 (2x4 WMMA frags), K-step 32.
// Double-buffered LDS staged with global_load_async_to_lds_b128 (ASYNCcnt pipelining).
// M must be a multiple of 128; K a multiple of 32; N guarded.
__global__ __launch_bounds__(256) void gemm_bf16_wmma(
    const bf16* __restrict__ A, const bf16* __restrict__ W,
    float* __restrict__ Cf, bf16* __restrict__ Cb,
    int M, int N, int K)
{
  __shared__ __align__(16) bf16 As[2][128 * 32];
  __shared__ __align__(16) bf16 Bs[2][128 * 32];

  const int tid  = threadIdx.x;
  const int lane = tid & 31;
  const int wid  = tid >> 5;
  const int wm   = wid & 3;        // 4 wave rows   -> 32 M-rows each
  const int wn   = wid >> 2;       // 2 wave cols   -> 64 N-cols each
  const int lh   = lane & 15;
  const int khalf= lane >> 4;
  const int m0   = blockIdx.y * 128;
  const int n0   = blockIdx.x * 128;
  const int arow = tid >> 1;             // 0..127
  const int aseg = (tid & 1) * 16;       // 0 or 16 halfs

  const bool fullN = (n0 + 128 <= N);    // uniform across the block

  floatx8 acc[2][4];
#pragma unroll
  for (int mi = 0; mi < 2; ++mi)
#pragma unroll
    for (int ni = 0; ni < 4; ++ni) acc[mi][ni] = zero8();

  // stage one K-tile (k0) into LDS buffer `buf`
  auto stage = [&](int k0, int buf) {
    bf16* Asb = &As[buf][0];
    bf16* Bsb = &Bs[buf][0];
    const bf16* asrc = A + (long)(m0 + arow) * K + k0 + aseg;
    async_b128(&Asb[arow * 32 + aseg],     asrc);       // 2 async per wave for A
    async_b128(&Asb[arow * 32 + aseg + 8], asrc + 8);
    if (fullN) {
      const bf16* bsrc = W + (long)(n0 + arow) * K + k0 + aseg;
      async_b128(&Bsb[arow * 32 + aseg],     bsrc);     // +2 async per wave for B
      async_b128(&Bsb[arow * 32 + aseg + 8], bsrc + 8);
    } else {
      // edge block: zero-pad rows beyond N (async path cannot write zeros)
      uint4 v0 = {0u,0u,0u,0u}, v1 = {0u,0u,0u,0u};
      if (n0 + arow < N) {
        const bf16* bsrc = W + (long)(n0 + arow) * K + k0 + aseg;
        v0 = *(const uint4*)bsrc;
        v1 = *(const uint4*)(bsrc + 8);
      }
      *(uint4*)&Bsb[arow * 32 + aseg]     = v0;
      *(uint4*)&Bsb[arow * 32 + aseg + 8] = v1;
    }
  };

  stage(0, 0);
  int cur = 0;

  for (int k0 = 0; k0 < K; k0 += 32) {
    // pipeline: issue next tile, then wait for ONLY the current tile
    // (async loads retire in order; per-wave per-tile count = 4 (fullN) / 2).
    if (k0 + 32 < K) {
      stage(k0 + 32, cur ^ 1);
      if (fullN) wait_async<4>(); else wait_async<2>();
    } else {
      wait_async<0>();
    }
    __syncthreads();                       // current tile visible to all waves

    const bf16* Asb = &As[cur][0];
    const bf16* Bsb = &Bs[cur][0];
    bf16x16 af[2], bfrag[4];
#pragma unroll
    for (int mi = 0; mi < 2; ++mi) {
      // A 16x32 layout: lanes 0-15 rows, K chunks at khalf*8 and 16+khalf*8
      const bf16* p = &Asb[(wm * 32 + mi * 16 + lh) * 32 + khalf * 8];
      af[mi] = ldfrag(p, p + 16);
    }
#pragma unroll
    for (int ni = 0; ni < 4; ++ni) {
      // B 32x16 layout: K chunks at khalf*8 and 16+khalf*8 (matches A)
      const bf16* p = &Bsb[(wn * 64 + ni * 16 + lh) * 32 + khalf * 8];
      bfrag[ni] = ldfrag(p, p + 16);
    }
#pragma unroll
    for (int mi = 0; mi < 2; ++mi)
#pragma unroll
      for (int ni = 0; ni < 4; ++ni)
        acc[mi][ni] = wmma_bf(af[mi], bfrag[ni], acc[mi][ni]);

    __syncthreads();                       // all reads done before buffer reuse
    cur ^= 1;
  }

  // store (C layout: lane holds col lh; rows j + khalf*8)
#pragma unroll
  for (int mi = 0; mi < 2; ++mi) {
    const int rbase = m0 + wm * 32 + mi * 16 + khalf * 8;
#pragma unroll
    for (int ni = 0; ni < 4; ++ni) {
      const int col = n0 + wn * 64 + ni * 16 + lh;
      if (col < N) {
#pragma unroll
        for (int j = 0; j < 8; ++j) {
          const long o = (long)(rbase + j) * N + col;
          const float v = acc[mi][ni][j];
          if (Cf) Cf[o] = v; else Cb[o] = f2bf(v);
        }
      }
    }
  }
}

// ---------------- RMSNorm: out(bf16) = in * rsqrt(mean(in^2)+eps) * g ----------------
__global__ __launch_bounds__(256) void rmsnorm_k(const float* __restrict__ in,
                                                 const float* __restrict__ g,
                                                 bf16* __restrict__ out, int N)
{
  const int row = blockIdx.x, tid = threadIdx.x;
  const float* p = in + (long)row * N;
  float s = 0.f;
  for (int i = tid; i < N; i += 256) { float v = p[i]; s += v * v; }
#pragma unroll
  for (int m2 = 16; m2 >= 1; m2 >>= 1) s += __shfl_xor(s, m2, 32);
  __shared__ float red[8];
  if ((tid & 31) == 0) red[tid >> 5] = s;
  __syncthreads();
  if (tid == 0) {
    float t = 0.f;
#pragma unroll
    for (int i = 0; i < 8; ++i) t += red[i];
    red[0] = rsqrtf(t / (float)N + 1e-6f);
  }
  __syncthreads();
  const float sc = red[0];
  bf16* q = out + (long)row * N;
  for (int i = tid; i < N; i += 256) q[i] = f2bf(p[i] * sc * g[i]);
}

// ---------------- assemble Q = [q_C | rope(q_R)] into [B*H, S, 576] ----------------
__global__ __launch_bounds__(256) void assemble_q(const bf16* __restrict__ qc,
                                                  const bf16* __restrict__ qrm,
                                                  const float* __restrict__ cosb,
                                                  const float* __restrict__ sinb,
                                                  bf16* __restrict__ qh)
{
  int idx = blockIdx.x * 256 + threadIdx.x;
  if (idx >= B_ * H_ * S_ * QKD_) return;
  int d = idx % QKD_; int t = idx / QKD_;
  int s = t % S_; t /= S_;
  int h = t % H_; int b = t / H_;
  bf16 val;
  if (d < DH_) {
    val = qc[(long)(b * S_ + s) * HD_ + h * DH_ + d];
  } else {
    int i = d - DH_, p = i >> 1;
    float c  = cosb[s * (RD_/2) + p];
    float sn = sinb[s * (RD_/2) + p];
    const long base = (long)(b * S_ + s) * HR_ + h * RD_ + 2 * p;
    float xr = bf2f(qrm[base]), xi = bf2f(qrm[base + 1]);
    val = f2bf((i & 1) ? (xr * sn + xi * c) : (xr * c - xi * sn));
  }
  qh[((long)(b * H_ + h) * S_ + s) * QKD_ + d] = val;
}

// ---------------- assemble K = [k_C | broadcast rope(k_R)] into [B*H, S, 576] --------
__global__ __launch_bounds__(256) void assemble_k(const bf16* __restrict__ kc,
                                                  const bf16* __restrict__ krm,
                                                  const float* __restrict__ cosb,
                                                  const float* __restrict__ sinb,
                                                  bf16* __restrict__ kh)
{
  int idx = blockIdx.x * 256 + threadIdx.x;
  if (idx >= B_ * H_ * S_ * QKD_) return;
  int d = idx % QKD_; int t = idx / QKD_;
  int s = t % S_; t /= S_;
  int h = t % H_; int b = t / H_;
  bf16 val;
  if (d < DH_) {
    val = kc[(long)(b * S_ + s) * HD_ + h * DH_ + d];
  } else {
    int i = d - DH_, p = i >> 1;
    float c  = cosb[s * (RD_/2) + p];
    float sn = sinb[s * (RD_/2) + p];
    const long base = (long)(b * S_ + s) * RD_ + 2 * p;  // shared across heads
    float xr = bf2f(krm[base]), xi = bf2f(krm[base + 1]);
    val = f2bf((i & 1) ? (xr * sn + xi * c) : (xr * c - xi * sn));
  }
  kh[((long)(b * H_ + h) * S_ + s) * QKD_ + d] = val;
}

// ---------------- causal flash attention (WMMA, online softmax) ----------------
// grid: (S/64, B*H); block 256 (8 waves): 4 q-row-groups x 2 V-column halves.
// dynamic LDS: Ks 32x576 (36864B) + Vs^T 384x32 (24576B) + Ps 4x16x32 (4096B) = 65536B
// K tile staged with async global->LDS DMA; V staged manually (transpose scatter).
__global__ __launch_bounds__(256) void mla_attn(const bf16* __restrict__ Qh,
                                                const bf16* __restrict__ Kh,
                                                const bf16* __restrict__ Vv,
                                                bf16* __restrict__ Ob, float scale)
{
  extern __shared__ char smem[];
  bf16* Ks = (bf16*)smem;                       // [32][576]
  bf16* Vs = (bf16*)(smem + 36864);             // transposed [384][32]
  bf16* Ps = (bf16*)(smem + 36864 + 24576);     // [4][16][32]

  const int tid = threadIdx.x, lane = tid & 31, wid = tid >> 5;
  const int rg = wid & 3;       // q-row group (16 rows)
  const int ch = wid >> 2;      // V column half (192 cols)
  const int bh = blockIdx.y, b = bh >> 4, h = bh & 15;
  const int q0 = blockIdx.x * 64;
  const int lh = lane & 15, khalf = lane >> 4;
  const int rowq = q0 + rg * 16;

  // Q fragments for this wave's 16 rows: 18 K-steps of 32 over 576, kept in VGPRs
  bf16x16 qf[18];
  {
    const bf16* qbase = Qh + ((long)bh * S_ + rowq + lh) * QKD_;
#pragma unroll
    for (int ks = 0; ks < 18; ++ks) {
      const bf16* p = qbase + ks * 32 + khalf * 8;   // A-matrix 16x32 layout
      qf[ks] = ldfrag(p, p + 16);
    }
  }

  floatx8 oacc[12];
#pragma unroll
  for (int ni = 0; ni < 12; ++ni) oacc[ni] = zero8();
  float mstate[8], lstate[8];
#pragma unroll
  for (int j = 0; j < 8; ++j) { mstate[j] = -1e30f; lstate[j] = 0.f; }

  const int kend = q0 + 64;                         // causal upper bound
  for (int k0 = 0; k0 < kend; k0 += 32) {
    __syncthreads();
    // stage K tile [32][576] via async DMA to LDS (9 instr/wave)
    for (int c = tid; c < 2304; c += 256) {
      int r = c / 72, off = (c % 72) * 8;
      async_b128(&Ks[r * 576 + off],
                 &Kh[((long)bh * S_ + k0 + r) * QKD_ + off]);
    }
    // stage V tile transposed [384][32] (manual: needs scatter)
    for (int c = tid; c < 1536; c += 256) {
      int r = c / 48, off = (c % 48) * 8;
      uint4 tmp = *(const uint4*)&Vv[((long)(b * S_) + k0 + r) * HD_ + h * DH_ + off];
      const bf16* pv = (const bf16*)&tmp;
#pragma unroll
      for (int t2 = 0; t2 < 8; ++t2) Vs[(off + t2) * 32 + r] = pv[t2];
    }
    wait_async<0>();
    __syncthreads();

    // scores S = Q.K^T for 16 q-rows x 32 keys
    floatx8 sc[2];
#pragma unroll
    for (int nt = 0; nt < 2; ++nt) {
      floatx8 a = zero8();
      const bf16* kb = &Ks[(nt * 16 + lh) * 576];
#pragma unroll
      for (int ks = 0; ks < 18; ++ks) {
        const bf16* p = kb + ks * 32 + khalf * 8;    // B-matrix 32x16 layout
        a = wmma_bf(qf[ks], ldfrag(p, p + 16), a);
      }
      sc[nt] = a;
    }

    // online softmax update (rows j + khalf*8, col = lh within each 16-tile)
    float pv2[2][8];
#pragma unroll
    for (int j = 0; j < 8; ++j) {
      const int qr = rowq + j + khalf * 8;
      float a0 = sc[0][j] * scale;
      float a1 = sc[1][j] * scale;
      if (k0 + lh > qr)      a0 = -__builtin_inff();
      if (k0 + 16 + lh > qr) a1 = -__builtin_inff();
      float rmax = fmaxf(a0, a1);
#pragma unroll
      for (int m2 = 8; m2 >= 1; m2 >>= 1) rmax = fmaxf(rmax, __shfl_xor(rmax, m2, 32));
      const float mnew = fmaxf(mstate[j], rmax);
      const float corr = __expf(mstate[j] - mnew);
      const float p0 = __expf(a0 - mnew);
      const float p1 = __expf(a1 - mnew);
      float rsum = p0 + p1;
#pragma unroll
      for (int m2 = 8; m2 >= 1; m2 >>= 1) rsum += __shfl_xor(rsum, m2, 32);
      lstate[j] = lstate[j] * corr + rsum;
      mstate[j] = mnew;
      pv2[0][j] = p0; pv2[1][j] = p1;
#pragma unroll
      for (int ni = 0; ni < 12; ++ni) oacc[ni][j] *= corr;
    }

    // transpose P through LDS into A-fragment layout (only ch==0 wave writes)
    if (ch == 0) {
#pragma unroll
      for (int nt = 0; nt < 2; ++nt)
#pragma unroll
        for (int j = 0; j < 8; ++j) {
          const int m = j + khalf * 8;
          Ps[rg * 512 + m * 32 + nt * 16 + lh] = f2bf(pv2[nt][j]);
        }
    }
    __syncthreads();

    // O += P(16x32) x V(32x192-half)
    const bf16* pb = &Ps[rg * 512 + lh * 32 + khalf * 8];
    bf16x16 pf = ldfrag(pb, pb + 16);
#pragma unroll
    for (int ni = 0; ni < 12; ++ni) {
      const bf16* vb = &Vs[(ch * 192 + ni * 16 + lh) * 32 + khalf * 8];
      oacc[ni] = wmma_bf(pf, ldfrag(vb, vb + 16), oacc[ni]);
    }
  }

  // epilogue: O /= l, write to [token, h*384 + col] bf16
#pragma unroll
  for (int j = 0; j < 8; ++j) {
    const int qr = rowq + j + khalf * 8;
    const float inv = 1.0f / lstate[j];
#pragma unroll
    for (int ni = 0; ni < 12; ++ni) {
      Ob[((long)(b * S_) + qr) * HD_ + h * DH_ + ch * 192 + ni * 16 + lh] =
          f2bf(oacc[ni][j] * inv);
    }
  }
}

// ---------------- launcher ----------------
extern "C" void kernel_launch(void* const* d_in, const int* in_sizes, int n_in,
                              void* d_out, int out_size, void* d_ws, size_t ws_size,
                              hipStream_t stream) {
  (void)in_sizes; (void)n_in; (void)out_size; (void)ws_size;
  const float* x    = (const float*)d_in[0];
  const float* WDQ  = (const float*)d_in[1];
  const float* WUQ  = (const float*)d_in[2];
  const float* WDWK = (const float*)d_in[3];
  const float* WUK  = (const float*)d_in[4];
  const float* WUV  = (const float*)d_in[5];
  const float* WQR  = (const float*)d_in[6];
  const float* WKR  = (const float*)d_in[7];
  const float* WO   = (const float*)d_in[8];
  const float* g_q  = (const float*)d_in[9];
  const float* g_kv = (const float*)d_in[10];
  const float* cosb = (const float*)d_in[11];
  const float* sinb = (const float*)d_in[12];
  float* out = (float*)d_out;

  // bump allocator over d_ws (~700 MB total)
  char* ws = (char*)d_ws;
  size_t off = 0;
  auto alloc = [&](size_t bytes) -> char* {
    char* p = ws + off;
    off += (bytes + 255) & ~(size_t)255;
    return p;
  };
  auto abf = [&](long n) -> bf16* { return (bf16*)alloc((size_t)n * 2); };

  bf16* xb   = abf((long)MTOK * D_);
  bf16* wdq  = abf((long)DCQ_ * D_);
  bf16* wuq  = abf((long)HD_ * DCQ_);
  bf16* wdwk = abf((long)DCKV_ * D_);
  bf16* wuk  = abf((long)HD_ * DCKV_);
  bf16* wuv  = abf((long)HD_ * DCKV_);
  bf16* wqr  = abf((long)HR_ * DCQ_);
  bf16* wkr  = abf((long)RD_ * D_);
  bf16* wo   = abf((long)D_ * HD_);
  float* tmpf = (float*)alloc((size_t)MTOK * DCQ_ * 4);   // reused f32 scratch
  bf16* cqb  = abf((long)MTOK * DCQ_);
  bf16* ckvb = abf((long)MTOK * DCKV_);
  bf16* qc   = abf((long)MTOK * HD_);
  bf16* kc   = abf((long)MTOK * HD_);
  bf16* vv   = abf((long)MTOK * HD_);
  bf16* qr   = abf((long)MTOK * HR_);
  bf16* kr   = abf((long)MTOK * RD_);
  bf16* qh   = abf((long)B_ * H_ * S_ * QKD_);
  bf16* kh   = abf((long)B_ * H_ * S_ * QKD_);
  bf16* obf  = abf((long)MTOK * HD_);

  auto cvt = [&](const float* s, bf16* d, long n) {
    cvt_f32_bf16<<<dim3((unsigned)((n + 255) / 256)), 256, 0, stream>>>(s, d, n);
  };
  auto gemm = [&](const bf16* A, const bf16* W, float* Cf, bf16* Cb,
                  int M, int N, int K) {
    dim3 g((N + 127) / 128, M / 128);
    gemm_bf16_wmma<<<g, 256, 0, stream>>>(A, W, Cf, Cb, M, N, K);
  };

  // 1) precision conversion
  cvt(x,    xb,   (long)MTOK * D_);
  cvt(WDQ,  wdq,  (long)DCQ_ * D_);
  cvt(WUQ,  wuq,  (long)HD_ * DCQ_);
  cvt(WDWK, wdwk, (long)DCKV_ * D_);
  cvt(WUK,  wuk,  (long)HD_ * DCKV_);
  cvt(WUV,  wuv,  (long)HD_ * DCKV_);
  cvt(WQR,  wqr,  (long)HR_ * DCQ_);
  cvt(WKR,  wkr,  (long)RD_ * D_);
  cvt(WO,   wo,   (long)D_ * HD_);

  // 2) down-projections + RMSNorm
  gemm(xb, wdq, tmpf, nullptr, MTOK, DCQ_, D_);
  rmsnorm_k<<<MTOK, 256, 0, stream>>>(tmpf, g_q, cqb, DCQ_);
  gemm(xb, wdwk, tmpf, nullptr, MTOK, DCKV_, D_);
  rmsnorm_k<<<MTOK, 256, 0, stream>>>(tmpf, g_kv, ckvb, DCKV_);

  // 3) up-projections (bf16 out)
  gemm(cqb,  wuq, nullptr, qc, MTOK, HD_, DCQ_);
  gemm(ckvb, wuk, nullptr, kc, MTOK, HD_, DCKV_);
  gemm(ckvb, wuv, nullptr, vv, MTOK, HD_, DCKV_);
  gemm(cqb,  wqr, nullptr, qr, MTOK, HR_, DCQ_);
  gemm(xb,   wkr, nullptr, kr, MTOK, RD_, D_);

  // 4) RoPE + concat into per-head layouts
  {
    const int total = B_ * H_ * S_ * QKD_;
    assemble_q<<<(total + 255) / 256, 256, 0, stream>>>(qc, qr, cosb, sinb, qh);
    assemble_k<<<(total + 255) / 256, 256, 0, stream>>>(kc, kr, cosb, sinb, kh);
  }

  // 5) causal flash attention
  {
    const float scale = 0.041666666666666664f; // (DH+RD)^-0.5 = 1/24
    dim3 g(S_ / 64, B_ * H_);
    mla_attn<<<g, 256, 65536, stream>>>(qh, kh, vv, obf, scale);
  }

  // 6) output projection -> f32 d_out
  gemm(obf, wo, out, nullptr, MTOK, D_, HD_);
}
